// GCN3D_Apr14_PoolingNoFc_66151086293371
// MI455X (gfx1250) — matrix operand order repaired and checked by reading, not compiled, not run on hardware
//
#include <hip/hip_runtime.h>
#include <hip/hip_bf16.h>

// ---------------- problem constants (from reference) ----------------
#define NODES   80000      // N = B*NPG (multiple of 32)
#define NGRAPH  8          // B
#define NPG     10000
#define KCLUS   512        // K
#define CLUS    4096       // C = B*K (multiple of 32)
#define NFEAT   16
#define FWIDE   256

#define PAD32(x) (((x) + 31) & ~31)

typedef __attribute__((ext_vector_type(16))) _Float16 v16h;
typedef __attribute__((ext_vector_type(8)))  _Float16 v8h;
typedef __attribute__((ext_vector_type(8)))  float    v8f;

__device__ __forceinline__ float elu1(float v) { return v > 0.f ? v : (__expf(v) - 1.f); }

// =====================================================================
// WMMA GEMM on pre-staged f16 operands.
//   A : f16 row-major [ceil32(M), Kp]   (zero-padded)
//   Bt: f16 row-major [ceil32(N), Kp]   (= B^T, zero-padded)
//   C : f32 [M, N] tight (+bias, +ELU)
// Each wave32 computes a 32x32 output supertile = 2x2 WMMA tiles.
// Inner loop: 8 x b128 loads + 4 x v_wmma_f32_16x16x32_f16, no guards.
// All branches are wave-uniform -> EXEC all-1s at every WMMA.
// =====================================================================
__global__ __launch_bounds__(256)
void k_gemm16(const _Float16* __restrict__ A, const _Float16* __restrict__ Bt,
              float* __restrict__ C, int M, int N, int Kp,
              const float* __restrict__ bias, int act)
{
    const int lane = threadIdx.x & 31;
    const int gw   = (int)((blockIdx.x * (unsigned)blockDim.x + threadIdx.x) >> 5);
    const int ntn  = (N + 31) >> 5;
    const int ntm  = (M + 31) >> 5;
    if (gw >= ntm * ntn) return;                   // wave-uniform exit
    const int tm = gw / ntn, tn = gw % ntn;
    const int l15 = lane & 15;
    const int hi  = lane >> 4;                     // half-wave select
    const int aoff = hi * 16;                      // A K sub-offset (contiguous, matches B)
    const int boff = hi * 16;                      // B K sub-offset (contiguous, matches A)

    const _Float16* pa0 = A  + (long)(tm * 32 + l15)      * Kp + aoff;
    const _Float16* pa1 = pa0 + (long)16 * Kp;
    const _Float16* pb0 = Bt + (long)(tn * 32 + l15)      * Kp + boff;
    const _Float16* pb1 = pb0 + (long)16 * Kp;

    v8f acc00 = {}, acc01 = {}, acc10 = {}, acc11 = {};
    for (int kc = 0; kc < Kp; kc += 32) {
        if (kc + 32 < Kp) {                        // prefetch next K chunk
            __builtin_prefetch(pa0 + kc + 32, 0, 1);
            __builtin_prefetch(pb0 + kc + 32, 0, 1);
        }
        // A fragments: two 16B-aligned v8h pieces (K = aoff..+7, aoff+8..+15)
        v8h a0lo = *(const v8h*)(pa0 + kc),  a0hi = *(const v8h*)(pa0 + kc + 8);
        v8h a1lo = *(const v8h*)(pa1 + kc),  a1hi = *(const v8h*)(pa1 + kc + 8);
        v16h a0 = __builtin_shufflevector(a0lo, a0hi, 0,1,2,3,4,5,6,7,8,9,10,11,12,13,14,15);
        v16h a1 = __builtin_shufflevector(a1lo, a1hi, 0,1,2,3,4,5,6,7,8,9,10,11,12,13,14,15);
        // B fragments: one 32B-aligned contiguous v16h each (K = boff..boff+15)
        v16h b0 = *(const v16h*)(pb0 + kc);
        v16h b1 = *(const v16h*)(pb1 + kc);

        acc00 = __builtin_amdgcn_wmma_f32_16x16x32_f16(false, a0, false, b0, (short)0, acc00, false, false);
        acc01 = __builtin_amdgcn_wmma_f32_16x16x32_f16(false, a0, false, b1, (short)0, acc01, false, false);
        acc10 = __builtin_amdgcn_wmma_f32_16x16x32_f16(false, a1, false, b0, (short)0, acc10, false, false);
        acc11 = __builtin_amdgcn_wmma_f32_16x16x32_f16(false, a1, false, b1, (short)0, acc11, false, false);
    }

    // D layout: VGPR r -> M = r (lanes 0-15) / r+8 (lanes 16-31), N = lane%16
    const int rb = tm * 32 + hi * 8;
    const int c0 = tn * 32 + l15, c1 = c0 + 16;
#pragma unroll
    for (int r = 0; r < 8; ++r) {
        int r0 = rb + r, r1 = r0 + 16;
        float v;
        if (r0 < M && c0 < N) { v = acc00[r]; if (bias) v += bias[c0]; if (act) v = elu1(v); C[(long)r0*N + c0] = v; }
        if (r0 < M && c1 < N) { v = acc01[r]; if (bias) v += bias[c1]; if (act) v = elu1(v); C[(long)r0*N + c1] = v; }
        if (r1 < M && c0 < N) { v = acc10[r]; if (bias) v += bias[c0]; if (act) v = elu1(v); C[(long)r1*N + c0] = v; }
        if (r1 < M && c1 < N) { v = acc11[r]; if (bias) v += bias[c1]; if (act) v = elu1(v); C[(long)r1*N + c1] = v; }
    }
}

// ---------------- f16 staging kernels --------------------------------
// row-major convert+pad: src f32 [M,K] -> dst f16 [Mp,Kp], zero padding
__global__ void k_cvt_pad(const float* __restrict__ src, _Float16* __restrict__ dst,
                          int M, int K, int Mp, int Kp) {
    long t = blockIdx.x * (long)blockDim.x + threadIdx.x;
    long tot = (long)Mp * Kp;
    if (t >= tot) return;
    int r = (int)(t / Kp), k = (int)(t % Kp);
    dst[t] = (r < M && k < K) ? (_Float16)src[(long)r * K + k] : (_Float16)0.f;
}

// tiled LDS transpose+convert+pad: src f32 [K,N] -> dst f16 [Np,Kp] (dst[n,k]=src[k,n]*scale[k])
__global__ __launch_bounds__(256)
void k_cvt_t_pad(const float* __restrict__ src, _Float16* __restrict__ dst,
                 int K, int N, int Kp, int Np, const float* __restrict__ scale) {
    __shared__ _Float16 tile[32][34];
    int kb = blockIdx.x * 32;                   // K block
    int nb = blockIdx.y * 32;                   // N block
    int tx = threadIdx.x & 31, ty = threadIdx.x >> 5;
    for (int i = ty; i < 32; i += 8) {
        int k = kb + i, n = nb + tx;
        float v = (k < K && n < N) ? src[(long)k * N + n] : 0.f;
        if (scale && k < K) v *= scale[k];
        tile[i][tx] = (_Float16)v;
    }
    __syncthreads();
    for (int i = ty; i < 32; i += 8) {
        int n = nb + i, k = kb + tx;
        if (n < Np && k < Kp) dst[(long)n * Kp + k] = tile[tx][i];
    }
}

// ------------------- sparse-graph helper kernels ---------------------
__global__ void k_deg(const int* __restrict__ dst, float* __restrict__ deg, int E) {
    int e = blockIdx.x * blockDim.x + threadIdx.x;
    if (e < E) atomicAdd(&deg[dst[e]], 1.0f);
}

__global__ void k_dinv(const float* __restrict__ deg, float* __restrict__ dinv, int n) {
    int i = blockIdx.x * blockDim.x + threadIdx.x;
    if (i < n) dinv[i] = rsqrtf(deg[i] + 1.0f);
}

__global__ void k_gcn_init(const float* __restrict__ h, const float* __restrict__ dinv,
                           const float* __restrict__ bias, float* __restrict__ out, int F) {
    long t = blockIdx.x * (long)blockDim.x + threadIdx.x;
    long tot = (long)NODES * F;
    if (t >= tot) return;
    int r = (int)(t / F), c = (int)(t % F);
    float di = dinv[r];
    out[t] = h[t] * di * di + bias[c];
}

__global__ void k_edge_scatter(const float* __restrict__ h, const int* __restrict__ src,
                               const int* __restrict__ dst, const float* __restrict__ dinv,
                               float* __restrict__ out, int E, int F) {
    long t = blockIdx.x * (long)blockDim.x + threadIdx.x;
    int per = F >> 2;
    long tot = (long)E * per;
    if (t >= tot) return;
    int e  = (int)(t / per);
    int f4 = (int)(t % per) * 4;
    int s = src[e], d = dst[e];
    float w = dinv[s] * dinv[d];
    const float4 v = *(const float4*)(h + (long)s * F + f4);
    float* o = out + (long)d * F + f4;
    atomicAdd(o + 0, v.x * w);
    atomicAdd(o + 1, v.y * w);
    atomicAdd(o + 2, v.z * w);
    atomicAdd(o + 3, v.w * w);
}

__global__ void k_elu_ip(float* __restrict__ p, long n) {
    long t = blockIdx.x * (long)blockDim.x + threadIdx.x;
    if (t < n) p[t] = elu1(p[t]);
}

__global__ void k_inorm(float* __restrict__ h, int F) {
    int b = blockIdx.x / F, c = blockIdx.x % F;
    __shared__ float s1[256], s2[256];
    const long base = (long)b * NPG * F + c;
    float sum = 0.f, sq = 0.f;
    for (int i = threadIdx.x; i < NPG; i += 256) {
        float v = h[base + (long)i * F];
        sum += v; sq += v * v;
    }
    s1[threadIdx.x] = sum; s2[threadIdx.x] = sq;
    __syncthreads();
    for (int st = 128; st > 0; st >>= 1) {
        if (threadIdx.x < st) { s1[threadIdx.x] += s1[threadIdx.x + st];
                                s2[threadIdx.x] += s2[threadIdx.x + st]; }
        __syncthreads();
    }
    float mu  = s1[0] / (float)NPG;
    float var = s2[0] / (float)NPG - mu * mu;
    float rs  = rsqrtf(var + 1e-5f);
    for (int i = threadIdx.x; i < NPG; i += 256) {
        long idx = base + (long)i * F;
        h[idx] = (h[idx] - mu) * rs;
    }
}

// ------------------------- pooling kernels ---------------------------
__global__ void k_cnt(const int* __restrict__ inb, const int* __restrict__ clus,
                      float* __restrict__ cnt, int n) {
    int i = blockIdx.x * blockDim.x + threadIdx.x;
    if (i < n) atomicAdd(&cnt[clus[i] + inb[i] * KCLUS], 1.0f);
}

__global__ void k_pool_accum(const float* __restrict__ y, const int* __restrict__ inb,
                             const int* __restrict__ clus, float* __restrict__ px) {
    long t = blockIdx.x * (long)blockDim.x + threadIdx.x;
    const int per = FWIDE / 4;
    long tot = (long)NODES * per;
    if (t >= tot) return;
    int n  = (int)(t / per);
    int f4 = (int)(t % per) * 4;
    int bc = clus[n] + inb[n] * KCLUS;
    const float4 v = *(const float4*)(y + (long)n * FWIDE + f4);
    float* o = px + (long)bc * FWIDE + f4;
    atomicAdd(o + 0, v.x);
    atomicAdd(o + 1, v.y);
    atomicAdd(o + 2, v.z);
    atomicAdd(o + 3, v.w);
}

__global__ void k_pool_div(float* __restrict__ px, const float* __restrict__ cnt) {
    long t = blockIdx.x * (long)blockDim.x + threadIdx.x;
    long tot = (long)CLUS * FWIDE;
    if (t >= tot) return;
    px[t] /= cnt[t / FWIDE];
}

// ----------------------- dense-graph kernels -------------------------
// adjacency built directly in f16 (0/1 exact); idempotent stores, diag excluded
__global__ void k_buildA16(const int* __restrict__ src, const int* __restrict__ dst,
                           const int* __restrict__ inb, const int* __restrict__ clus,
                           _Float16* __restrict__ A, int E) {
    int e = blockIdx.x * blockDim.x + threadIdx.x;
    if (e >= E) return;
    int s = src[e], d = dst[e];
    int cs = clus[s] + inb[s] * KCLUS;
    int cd = clus[d] + inb[d] * KCLUS;
    if (cs != cd) A[(long)cd * CLUS + cs] = (_Float16)1.0f;
}

__global__ void k_rowsumA16(const _Float16* __restrict__ A, float* __restrict__ dinvc) {
    __shared__ float s1[256];
    const _Float16* r = A + (long)blockIdx.x * CLUS;
    float sum = 0.f;
    for (int i = threadIdx.x; i < CLUS; i += 256) sum += (float)r[i];
    s1[threadIdx.x] = sum;
    __syncthreads();
    for (int st = 128; st > 0; st >>= 1) {
        if (threadIdx.x < st) s1[threadIdx.x] += s1[threadIdx.x + st];
        __syncthreads();
    }
    if (threadIdx.x == 0) dinvc[blockIdx.x] = rsqrtf(s1[0] + 1.0f);
}

// out = elu( m*dinv + t*dinv^2 + bias )
__global__ void k_dense_combine(const float* __restrict__ m, const float* __restrict__ t,
                                const float* __restrict__ bias, const float* __restrict__ dinvc,
                                float* __restrict__ out, int F) {
    long i = blockIdx.x * (long)blockDim.x + threadIdx.x;
    long tot = (long)CLUS * F;
    if (i >= tot) return;
    int c = (int)(i % F);
    float di = dinvc[i / F];
    out[i] = elu1(m[i] * di + t[i] * di * di + bias[c]);
}

__global__ void k_copy(const float* __restrict__ a, float* __restrict__ b, long n) {
    long t = blockIdx.x * (long)blockDim.x + threadIdx.x;
    if (t < n) b[t] = a[t];
}

// ========================= host orchestration ========================
static inline void gemm16(const _Float16* A, const _Float16* Bt, float* C,
                          int M, int N, int Kp, const float* bias, int act,
                          hipStream_t s) {
    long waves  = (long)((M + 31) / 32) * ((N + 31) / 32);
    long blocks = (waves * 32 + 255) / 256;
    k_gemm16<<<(unsigned)blocks, 256, 0, s>>>(A, Bt, C, M, N, Kp, bias, act);
}

extern "C" void kernel_launch(void* const* d_in, const int* in_sizes, int n_in,
                              void* d_out, int out_size, void* d_ws, size_t ws_size,
                              hipStream_t stream)
{
    (void)n_in; (void)out_size;
    const float* x    = (const float*)d_in[0];
    const int*   adj  = (const int*)d_in[1];
    const int*   inb  = (const int*)d_in[2];
    const int*   clus = (const int*)d_in[3];
    const float *Wg1=(const float*)d_in[4],  *bg1=(const float*)d_in[5];
    const float *Wf1=(const float*)d_in[6],  *bf1=(const float*)d_in[7];
    const float *Wg2=(const float*)d_in[8],  *bg2=(const float*)d_in[9];
    const float *Wf2=(const float*)d_in[10], *bf2=(const float*)d_in[11];
    const float *Wl1=(const float*)d_in[12], *bl1=(const float*)d_in[13];
    const float *Wfl1=(const float*)d_in[14],*bfl1=(const float*)d_in[15];
    const float *Wl2=(const float*)d_in[16], *bl2=(const float*)d_in[17];
    const float *Wfl2=(const float*)d_in[18],*bfl2=(const float*)d_in[19];
    const float *Wm1=(const float*)d_in[20], *bm1=(const float*)d_in[21];
    const float *Wfm1=(const float*)d_in[22],*bfm1=(const float*)d_in[23];
    const float *Wo1=(const float*)d_in[24], *bo1=(const float*)d_in[25];
    const float *Wfo=(const float*)d_in[26];
    float* out = (float*)d_out;

    const int E = in_sizes[1] / 2;
    const int* src = adj;
    const int* dst = adj + E;

    // ---- workspace carve-out ----
    char* w = (char*)d_ws; size_t off = 0;
    auto takeb = [&](size_t bytes) -> void* {
        void* p = (void*)(w + off);
        off += (bytes + 255) & ~(size_t)255;
        return p;
    };
    float*     bufA  = (float*)takeb((size_t)NODES * FWIDE * 4);          // 80 MB
    float*     bufB  = (float*)takeb((size_t)NODES * FWIDE * 4);          // 80 MB
    _Float16*  Amat  = (_Float16*)takeb((size_t)CLUS * CLUS * 2);         // 32 MB
    _Float16*  a16   = (_Float16*)takeb((size_t)NODES * FWIDE * 2);       // 40 MB
    _Float16*  wfot  = (_Float16*)takeb((size_t)PAD32(3*NPG) * (5*KCLUS) * 2); // 154 MB
    float*     deg   = (float*)takeb((size_t)NODES * 4);
    float*     dinv  = (float*)takeb((size_t)NODES * 4);
    float*     dinvc = (float*)takeb((size_t)CLUS * 4);
    float*     cnt   = (float*)takeb((size_t)CLUS * 4);
    float*     sX    = (float*)takeb((size_t)CLUS * FWIDE * 4);
    float*     sY    = (float*)takeb((size_t)CLUS * FWIDE * 4);
    float*     sT    = (float*)takeb((size_t)CLUS * FWIDE * 4);
    float*     sM    = (float*)takeb((size_t)CLUS * FWIDE * 4);
    _Float16*  in16  = (_Float16*)takeb((size_t)CLUS * 256 * 2);
    _Float16*  hsT   = (_Float16*)takeb((size_t)128 * CLUS * 2);
    _Float16*  z16   = (_Float16*)takeb((size_t)32 * (5*KCLUS) * 2);
    _Float16*  wg1t  = (_Float16*)takeb((size_t)64  * 32  * 2);
    _Float16*  wf1t  = (_Float16*)takeb((size_t)128 * 64  * 2);
    _Float16*  wg2t  = (_Float16*)takeb((size_t)256 * 128 * 2);
    _Float16*  wf2t  = (_Float16*)takeb((size_t)256 * 256 * 2);
    _Float16*  wl1t  = (_Float16*)takeb((size_t)128 * 256 * 2);
    _Float16*  wfl1t = (_Float16*)takeb((size_t)96  * 128 * 2);
    _Float16*  wl2t  = (_Float16*)takeb((size_t)64  * 96  * 2);
    _Float16*  wfl2t = (_Float16*)takeb((size_t)32  * 64  * 2);
    _Float16*  wm1t  = (_Float16*)takeb((size_t)32  * 32  * 2);
    _Float16*  wfm1t = (_Float16*)takeb((size_t)32  * 32  * 2);
    _Float16*  wo1t  = (_Float16*)takeb((size_t)32  * 32  * 2);
    if (off > ws_size) return;                        // scratch too small: bail

    auto blk = [](long n) { return (unsigned)((n + 255) / 256); };
    auto cvt = [&](const float* s, _Float16* d, int M, int K, int Kp) {
        k_cvt_pad<<<blk((long)M * Kp), 256, 0, stream>>>(s, d, M, K, M, Kp);
    };
    auto cvtT = [&](const float* s, _Float16* d, int K, int N,
                    const float* scale) {  // -> [PAD32(N), PAD32(K)]
        dim3 g(PAD32(K) / 32, PAD32(N) / 32);
        k_cvt_t_pad<<<g, 256, 0, stream>>>(s, d, K, N, PAD32(K), PAD32(N), scale);
    };

    // ---- stage all weights transposed in f16 (once per launch) ----
    cvtT(Wg1,  wg1t,  16,  64,  nullptr);
    cvtT(Wf1,  wf1t,  64,  128, nullptr);
    cvtT(Wg2,  wg2t,  128, 256, nullptr);
    cvtT(Wf2,  wf2t,  256, 256, nullptr);
    cvtT(Wl1,  wl1t,  256, 128, nullptr);
    cvtT(Wfl1, wfl1t, 128, 96,  nullptr);
    cvtT(Wl2,  wl2t,  96,  64,  nullptr);
    cvtT(Wfl2, wfl2t, 64,  32,  nullptr);
    cvtT(Wm1,  wm1t,  32,  16,  nullptr);
    cvtT(Wfm1, wfm1t, 16,  8,   nullptr);
    cvtT(Wo1,  wo1t,  8,   5,   nullptr);
    cvtT(Wfo,  wfot,  5*KCLUS, 3*NPG, nullptr);

    // ---- node degrees / norm factors ----
    hipMemsetAsync(deg, 0, (size_t)NODES * 4, stream);
    k_deg<<<blk(E), 256, 0, stream>>>(dst, deg, E);
    k_dinv<<<blk(NODES), 256, 0, stream>>>(deg, dinv, NODES);

    // ---- sparse GCN layer 1: 16 -> 64 ----
    cvt(x, a16, NODES, NFEAT, 32);
    gemm16(a16, wg1t, bufA, NODES, 64, 32, nullptr, 0, stream);
    k_gcn_init<<<blk((long)NODES * 64), 256, 0, stream>>>(bufA, dinv, bg1, bufB, 64);
    k_edge_scatter<<<blk((long)E * 16), 256, 0, stream>>>(bufA, src, dst, dinv, bufB, E, 64);
    k_elu_ip<<<blk((long)NODES * 64), 256, 0, stream>>>(bufB, (long)NODES * 64);
    // fc1: 64 -> 128
    cvt(bufB, a16, NODES, 64, 64);
    gemm16(a16, wf1t, bufA, NODES, 128, 64, bf1, 1, stream);
    // ---- sparse GCN layer 2: 128 -> 256 ----
    cvt(bufA, a16, NODES, 128, 128);
    gemm16(a16, wg2t, bufB, NODES, 256, 128, nullptr, 0, stream);
    k_gcn_init<<<blk((long)NODES * 256), 256, 0, stream>>>(bufB, dinv, bg2, bufA, 256);
    k_edge_scatter<<<blk((long)E * 64), 256, 0, stream>>>(bufB, src, dst, dinv, bufA, E, 256);
    k_elu_ip<<<blk((long)NODES * 256), 256, 0, stream>>>(bufA, (long)NODES * 256);
    // fc2: 256 -> 256
    cvt(bufA, a16, NODES, 256, 256);
    gemm16(a16, wf2t, bufB, NODES, 256, 256, bf2, 1, stream);

    // ---- InstanceNorm per graph (in place, L2-resident second pass) ----
    k_inorm<<<NGRAPH * FWIDE, 256, 0, stream>>>(bufB, FWIDE);

    // ---- avg_pool to clusters -> px in sX ----
    hipMemsetAsync(cnt, 0, (size_t)CLUS * 4, stream);
    hipMemsetAsync(sX, 0, (size_t)CLUS * FWIDE * 4, stream);
    k_cnt<<<blk(NODES), 256, 0, stream>>>(inb, clus, cnt, NODES);
    k_pool_accum<<<blk((long)NODES * (FWIDE / 4)), 256, 0, stream>>>(bufB, inb, clus, sX);
    k_pool_div<<<blk((long)CLUS * FWIDE), 256, 0, stream>>>(sX, cnt);

    // ---- dense adjacency (f16) + its norm factors ----
    hipMemsetAsync(Amat, 0, (size_t)CLUS * CLUS * 2, stream);
    k_buildA16<<<blk(E), 256, 0, stream>>>(src, dst, inb, clus, Amat, E);
    k_rowsumA16<<<CLUS, 256, 0, stream>>>(Amat, dinvc);

    // ---- dense GCN stack ----
    auto dense_gcn = [&](const float* in, const _Float16* WT, const float* b,
                         int fin, int fout, float* o) {
        cvt(in, in16, CLUS, fin, PAD32(fin));
        gemm16(in16, WT, sT, CLUS, fout, PAD32(fin), nullptr, 0, stream);   // t = in @ W
        cvtT(sT, hsT, CLUS, fout, dinvc);                                   // hs^T (scaled)
        gemm16(Amat, hsT, sM, CLUS, fout, CLUS, nullptr, 0, stream);        // m = A @ hs
        k_dense_combine<<<blk((long)CLUS * fout), 256, 0, stream>>>(sM, sT, b, dinvc, o, fout);
    };
    auto fc = [&](const float* in, const _Float16* WT, const float* b,
                  int fin, int fout, float* o) {
        cvt(in, in16, CLUS, fin, PAD32(fin));
        gemm16(in16, WT, o, CLUS, fout, PAD32(fin), b, 1, stream);
    };

    dense_gcn(sX, wl1t, bl1, 256, 128, sY);
    fc(sY, wfl1t, bfl1, 128, 96, sX);
    dense_gcn(sX, wl2t, bl2, 96, 64, sY);
    fc(sY, wfl2t, bfl2, 64, 32, sX);
    dense_gcn(sX, wm1t, bm1, 32, 16, sY);
    fc(sY, wfm1t, bfm1, 16, 8, sX);
    dense_gcn(sX, wo1t, bo1, 8, 5, sY);                                     // z: [4096, 5]

    // ---- outputs: g_feat = z.flatten(); o_feat = z.reshape(8,2560) @ Wfo ----
    k_copy<<<blk((long)CLUS * 5), 256, 0, stream>>>(sY, out, (long)CLUS * 5);
    k_cvt_pad<<<blk((long)32 * (5*KCLUS)), 256, 0, stream>>>(sY, z16, NGRAPH, 5*KCLUS, 32, 5*KCLUS);
    gemm16(z16, wfot, out + (long)CLUS * 5, NGRAPH, 3 * NPG, 5 * KCLUS, nullptr, 0, stream);
}
